// CondConv2d_78194174591337
// MI455X (gfx1250) — compile-verified, matrix operand order checked by reading, not run on hardware
//
#include <hip/hip_runtime.h>

// ---------------------------------------------------------------------------
// CondConv2d for MI455X (gfx1250, wave32, WMMA + TDM).
// Compute-bound (AI ~525 flop/byte) -> bf16 v_wmma_f32_16x16x32_bf16 path.
// ---------------------------------------------------------------------------

typedef __attribute__((ext_vector_type(16))) __bf16          v16bf;
typedef __attribute__((ext_vector_type(8)))  __bf16          v8bf;
typedef __attribute__((ext_vector_type(8)))  float           v8f;
typedef __attribute__((ext_vector_type(4)))  unsigned int    v4u;
typedef __attribute__((ext_vector_type(8)))  int             v8i;
typedef __attribute__((ext_vector_type(4)))  int             v4i;

#if defined(__has_builtin)
#  if __has_builtin(__builtin_amdgcn_tensor_load_to_lds) && \
      __has_builtin(__builtin_amdgcn_s_wait_tensorcnt)
#    define HAS_TDM 1
#  else
#    define HAS_TDM 0
#  endif
#else
#  define HAS_TDM 0
#endif

// round-to-nearest-even float -> bf16 bits
static __device__ __forceinline__ unsigned short f2bf(float f) {
  unsigned u = __builtin_bit_cast(unsigned, f);
  u += 0x7FFFu + ((u >> 16) & 1u);
  return (unsigned short)(u >> 16);
}

// ---------------------------------------------------------------------------
// Kernel 1: global average pool + routing -> sigmoid gates  [n][8]
// ---------------------------------------------------------------------------
__global__ __launch_bounds__(256) void pool_routing_kernel(
    const float* __restrict__ x,      // [32][256][3136]
    const float* __restrict__ rw,     // [256][8]
    float* __restrict__ gates)        // [32][8]
{
  __shared__ float pooled[256];
  const int n    = blockIdx.x;
  const int tid  = threadIdx.x;
  const int lane = tid & 31;
  const int wid  = tid >> 5;

  for (int c = wid; c < 256; c += 8) {
    const float* xc = x + ((size_t)(n * 256 + c)) * 3136;
    float s = 0.f;
    for (int i = lane; i < 3136; i += 32) s += xc[i];
#pragma unroll
    for (int off = 16; off; off >>= 1) s += __shfl_xor(s, off, 32);
    if (lane == 0) pooled[c] = s * (1.0f / 3136.0f);
  }
  __syncthreads();
  if (tid < 8) {
    float z = 0.f;
    for (int c = 0; c < 256; ++c) z += pooled[c] * rw[c * 8 + tid];
    gates[n * 8 + tid] = 1.0f / (1.0f + __expf(-z));
  }
}

// ---------------------------------------------------------------------------
// Kernel 2: per-sample soft expert mix -> bf16 Wmix[n][oc][K], K = ic*9 = 2304
// ---------------------------------------------------------------------------
__global__ __launch_bounds__(256) void mix_weights_kernel(
    const float* __restrict__ weight,       // [8][256][2304]
    const float* __restrict__ gates,        // [32][8]
    unsigned short* __restrict__ Wmix)      // [32][256][2304] bf16
{
  const int oc = blockIdx.x;
  const int n  = blockIdx.y;
  float g[8];
#pragma unroll
  for (int e = 0; e < 8; ++e) g[e] = gates[n * 8 + e];
  unsigned short* dst = Wmix + ((size_t)(n * 256 + oc)) * 2304;
  for (int j = threadIdx.x; j < 2304; j += 256) {
    float acc = 0.f;
#pragma unroll
    for (int e = 0; e < 8; ++e)
      acc += g[e] * weight[((size_t)(e * 256 + oc)) * 2304 + j];
    dst[j] = f2bf(acc);
  }
}

// ---------------------------------------------------------------------------
// Kernel 3: pad (1,1,1,1) + convert x to bf16: xpad[n][c][58][58]
// ---------------------------------------------------------------------------
__global__ __launch_bounds__(256) void pad_convert_kernel(
    const float* __restrict__ x,            // [32][256][56][56]
    unsigned short* __restrict__ xpad)      // [32][256][58][58] bf16
{
  const int c = blockIdx.x;
  const int n = blockIdx.y;
  const float*   src = x    + ((size_t)(n * 256 + c)) * 3136;
  unsigned short* dst = xpad + ((size_t)(n * 256 + c)) * 3364;
  for (int idx = threadIdx.x; idx < 3364; idx += 256) {
    int row = idx / 58, col = idx - row * 58;
    unsigned short v = 0;
    if (row >= 1 && row <= 56 && col >= 1 && col <= 56)
      v = f2bf(src[(row - 1) * 56 + (col - 1)]);
    dst[idx] = v;
  }
}

// ---------------------------------------------------------------------------
// Kernel 4: implicit-GEMM conv per sample via WMMA bf16.
//   M = oc (256), N = pixels (3136), K = ic*9 (2304)
//   Workgroup: 128(M) x 128(N) tile, 8 waves in 4(M) x 2(N), K-step 64.
//   A tile DMA'd by the Tensor Data Mover into LDS (pad -> stride 72 halves),
//   B (im2col of xpad) staged in LDS [128][72] with division-free addressing.
// ---------------------------------------------------------------------------
__global__ __launch_bounds__(256) void condconv_wmma_kernel(
    const unsigned short* __restrict__ Wmix,   // [32][256][2304] bf16
    const unsigned short* __restrict__ xpad,   // [32][256][58][58] bf16
    float* __restrict__ out)                   // [32][256][3136]
{
  constexpr int LDST = 72;   // 64 K halves + 8 pad (stride 36 DW: bank-spread)
  __shared__ alignas(16) unsigned short smem[256 * LDST];
  unsigned short* As = smem;               // [128][72]  (LDS offset 0: TDM dst)
  unsigned short* Bs = smem + 128 * LDST;  // [128][72]

  const int n   = blockIdx.z;
  const int m0  = blockIdx.y * 128;   // oc tile base
  const int p0  = blockIdx.x * 128;   // pixel tile base
  const int tid = threadIdx.x;
  const int lane   = tid & 31;
  const int wid    = tid >> 5;
  const int waveM  = wid & 3;           // 4 x 32 oc rows
  const int waveN  = wid >> 2;          // 2 x 64 pixels
  const int lanelo = lane & 15;
  const int koff   = (lane >> 4) << 3;  // 0 or 8 (fragment K striping)

  const unsigned short* Wn = Wmix + (size_t)(n * 256) * 2304;
  const unsigned short* Xn = xpad + (size_t)(n * 256) * 3364;

  // ---- staging map: thread owns one k-lane (x2 reps) and 16 pixels ----
  const int klane  = tid >> 3;          // 0..31
  const int pchunk = (tid & 7) * 16;    // 0..112
  int offs[16];                          // xpad offsets (row*58+col), no halo
  unsigned pmask = 0;
  {
    int pb = p0 + pchunk;
    int py = pb / 56;                    // only divisions: once, pre-loop
    int px = pb - py * 56;
#pragma unroll
    for (int j = 0; j < 16; ++j) {
      if (pb + j < 3136) pmask |= (1u << j);
      offs[j] = py * 58 + px;
      ++px;
      if (px == 56) { px = 0; ++py; }
    }
  }

  v8f acc[2][4];
#pragma unroll
  for (int i = 0; i < 2; ++i)
#pragma unroll
    for (int j = 0; j < 4; ++j) acc[i][j] = v8f{};

  for (int kc = 0; kc < 2304; kc += 64) {
    __syncthreads();

#if HAS_TDM
    // ---- A tile via Tensor Data Mover: Wn[m0..m0+127][kc..kc+63] -> As ----
    if (wid == 0) {
      unsigned long long ga =
          (unsigned long long)(const void*)Wn + ((size_t)m0 * 2304 + kc) * 2;
      v4u g0;
      g0[0] = 1u;                                   // count=1 valid descriptor
      g0[1] = 0u;                                   // lds_addr = As (offset 0)
      g0[2] = (unsigned)ga;                         // global_addr[31:0]
      g0[3] = (unsigned)((ga >> 32) & 0x1FFFFFFull) // global_addr[56:32]
              | (2u << 30);                         // type = 2 ("image")
      v8i g1;
      g1[0] = (1 << 16)                             // data_size = 2 bytes
            | (1 << 20)                             // pad_enable
            | (4 << 22)                             // pad_interval: 32 DWORDs
            | (3 << 25);                            // pad_amount:   4 DWORDs
      g1[1] = (int)((2304u & 0xFFFFu) << 16);       // tensor_dim0 lo
      g1[2] = (int)((2304u >> 16) | (256u << 16));  // tensor_dim0 hi, dim1 lo
      g1[3] = (int)(64u << 16);                     // dim1 hi=0, tile_dim0=64
      g1[4] = 128;                                  // tile_dim1=128, tile_dim2=0
      g1[5] = 2304;                                 // tensor_dim0_stride lo32
      g1[6] = 0;
      g1[7] = 0;
      v4i z4 = {0, 0, 0, 0};
      v8i z8 = {0, 0, 0, 0, 0, 0, 0, 0};
      __builtin_amdgcn_tensor_load_to_lds(g0, g1, z4, z4, z8, 0);
    }
#endif

    // ---- stage B tile: Bs[p][k], one (ic,dy,dx) decode per thread/substep ----
#pragma unroll
    for (int krep = 0; krep < 2; ++krep) {
      int k  = kc + krep * 32 + klane;
      int ic = k / 9;
      int r  = k - ic * 9;
      int dy = r / 3;
      int dx = r - dy * 3;
      const unsigned short* src = Xn + ic * 3364 + dy * 58 + dx;
      unsigned short* dstB = &Bs[pchunk * LDST + krep * 32 + klane];
#pragma unroll
      for (int j = 0; j < 16; ++j) {
        unsigned short v = ((pmask >> j) & 1u) ? src[offs[j]] : (unsigned short)0;
        dstB[j * LDST] = v;
      }
      // prefetch next K-step's source region (~7 channels ahead)
      __builtin_prefetch(src + 7 * 3364 + offs[0], 0, 0);
    }

#if HAS_TDM
    if (wid == 0) __builtin_amdgcn_s_wait_tensorcnt((short)0);
#endif
    __syncthreads();

    // ---- two 32-K substeps of 2x4 WMMA ----
#pragma unroll
    for (int krep = 0; krep < 2; ++krep) {
      const int ks = krep * 32;

      v16bf a[2];
#pragma unroll
      for (int mt = 0; mt < 2; ++mt) {
#if HAS_TDM
        const unsigned short* ap =
            &As[(waveM * 32 + mt * 16 + lanelo) * LDST + ks + koff];
#else
        const unsigned short* ap =
            Wn + (size_t)(m0 + waveM * 32 + mt * 16 + lanelo) * 2304 + kc + ks + koff;
#endif
        v8bf lo = *(const v8bf*)ap;          // K 0..7   (lanes>=16: 8..15)
        v8bf hi = *(const v8bf*)(ap + 16);   // K 16..23 (lanes>=16: 24..31)
#pragma unroll
        for (int i = 0; i < 8; ++i) { a[mt][i] = lo[i]; a[mt][8 + i] = hi[i]; }
      }

      v16bf b[4];
#pragma unroll
      for (int nt = 0; nt < 4; ++nt) {
        const unsigned short* bp =
            &Bs[(waveN * 64 + nt * 16 + lanelo) * LDST + ks + koff];
        v8bf lo = *(const v8bf*)bp;
        v8bf hi = *(const v8bf*)(bp + 16);
#pragma unroll
        for (int i = 0; i < 8; ++i) { b[nt][i] = lo[i]; b[nt][8 + i] = hi[i]; }
      }

#pragma unroll
      for (int mt = 0; mt < 2; ++mt)
#pragma unroll
        for (int nt = 0; nt < 4; ++nt)
          acc[mt][nt] = __builtin_amdgcn_wmma_f32_16x16x32_bf16(
              false, a[mt], false, b[nt], (short)0, acc[mt][nt], false, false);
    }
  }

  // ---- epilogue: C layout: VGPR v -> M = base+v (lanes<16) / base+v+8 ----
  const int rowoff = koff;
#pragma unroll
  for (int mt = 0; mt < 2; ++mt) {
#pragma unroll
    for (int nt = 0; nt < 4; ++nt) {
      int col = p0 + waveN * 64 + nt * 16 + lanelo;
      if (col < 3136) {
        int rowbase = m0 + waveM * 32 + mt * 16 + rowoff;
        float* op = out + ((size_t)(n * 256 + rowbase)) * 3136 + col;
#pragma unroll
        for (int v = 0; v < 8; ++v)
          op[(size_t)v * 3136] = acc[mt][nt][v];
      }
    }
  }
}

// ---------------------------------------------------------------------------
// launch
// ---------------------------------------------------------------------------
extern "C" void kernel_launch(void* const* d_in, const int* in_sizes, int n_in,
                              void* d_out, int out_size, void* d_ws, size_t ws_size,
                              hipStream_t stream) {
  (void)in_sizes; (void)n_in; (void)out_size; (void)ws_size;
  const float* x      = (const float*)d_in[0];   // [32][256][56][56]
  const float* rw     = (const float*)d_in[1];   // [256][8]
  const float* weight = (const float*)d_in[2];   // [8][256][256][3][3]
  float*       out    = (float*)d_out;           // [32][256][56][56]

  // scratch: gates (1 KB) | Wmix bf16 (37.75 MB) | xpad bf16 (55.1 MB)
  float*          gates = (float*)d_ws;
  unsigned short* Wmix  = (unsigned short*)((char*)d_ws + 1024);
  unsigned short* xpad  = Wmix + (size_t)32 * 256 * 2304;

  pool_routing_kernel<<<32, 256, 0, stream>>>(x, rw, gates);
  mix_weights_kernel<<<dim3(256, 32), 256, 0, stream>>>(weight, gates, Wmix);
  pad_convert_kernel<<<dim3(256, 32), 256, 0, stream>>>(x, xpad);
  condconv_wmma_kernel<<<dim3(25, 2, 32), 256, 0, stream>>>(Wmix, xpad, out);
}
